// CrossScaleAttention_29910152250054
// MI455X (gfx1250) — matrix operand order repaired and checked by reading, not compiled, not run on hardware
//
#include <hip/hip_runtime.h>
#include <hip/hip_bf16.h>
#include <math.h>

// ---- problem constants ----
#define Bn   4
#define Cn   128
#define CRn  64
#define Hn   96
#define Wn   96
#define LHn  32           // low-res side
#define Ln   1024         // LHn*LHn patches
#define KDIM 576          // CRn*3*3
#define NPIX 9216         // Hn*Wn
#define OHW  288          // 3*Hn

typedef __bf16 v16bf_t __attribute__((ext_vector_type(16)));
typedef float  v8f_t   __attribute__((ext_vector_type(8)));

union FragU { v16bf_t v; uint4 u[2]; };

__device__ __forceinline__ unsigned short f2bf(float f) {
    unsigned int u = __float_as_uint(f);
    unsigned int r = (u + 0x7fffu + ((u >> 16) & 1u)) >> 16;
    return (unsigned short)r;
}

// ---------- zero pad row ----------
__global__ void k_zero(unsigned short* z) {
    z[blockIdx.x * blockDim.x + threadIdx.x] = 0;
}

// ---------- 1x1 conv + PReLU: embed (128ch) and match (64ch) ----------
__global__ void k_embed_match(const float* __restrict__ x,
                              const float* __restrict__ w_asm, const float* __restrict__ b_asm,
                              const float* __restrict__ a_asm,
                              const float* __restrict__ w_m1, const float* __restrict__ b_m1,
                              const float* __restrict__ a_m1,
                              float* __restrict__ embed, float* __restrict__ match) {
    __shared__ float xv[Cn];
    int p  = blockIdx.x;                 // b*H*W + h*W + w
    int b  = p / (Hn * Wn);
    int hw = p % (Hn * Wn);
    int t  = threadIdx.x;                // 0..127
    xv[t] = x[((size_t)(b * Cn + t)) * (Hn * Wn) + hw];
    __syncthreads();
    float acc = b_asm[t];
    const float* wr = w_asm + t * Cn;
    #pragma unroll 4
    for (int c = 0; c < Cn; ++c) acc = fmaf(wr[c], xv[c], acc);
    float a0 = a_asm[0];
    embed[((size_t)(b * Cn + t)) * (Hn * Wn) + hw] = acc >= 0.f ? acc : a0 * acc;
    if (t < CRn) {
        float acc2 = b_m1[t];
        const float* wr2 = w_m1 + t * Cn;
        #pragma unroll 4
        for (int c = 0; c < Cn; ++c) acc2 = fmaf(wr2[c], xv[c], acc2);
        float a1 = a_m1[0];
        match[((size_t)(b * CRn + t)) * (Hn * Wn) + hw] = acc2 >= 0.f ? acc2 : a1 * acc2;
    }
}

// ---------- exact bilinear(3x) downsample == point sample @ (3y+1,3x+1), then 1x1 conv ----------
__global__ void k_ref(const float* __restrict__ x,
                      const float* __restrict__ w_m2, const float* __restrict__ b_m2,
                      const float* __restrict__ a_m2,
                      float* __restrict__ ref) {
    __shared__ float xv[Cn];
    int p  = blockIdx.x;                 // b*32*32 + y*32 + xo
    int b  = p / (LHn * LHn);
    int yx = p % (LHn * LHn);
    int y  = yx / LHn, xo = yx % LHn;
    int t  = threadIdx.x;
    xv[t] = x[((size_t)(b * Cn + t)) * (Hn * Wn) + (3 * y + 1) * Wn + (3 * xo + 1)];
    __syncthreads();
    if (t < CRn) {
        float acc = b_m2[t];
        const float* wr = w_m2 + t * Cn;
        #pragma unroll 4
        for (int c = 0; c < Cn; ++c) acc = fmaf(wr[c], xv[c], acc);
        float a = a_m2[0];
        ref[((size_t)(b * CRn + t)) * (LHn * LHn) + yx] = acc >= 0.f ? acc : a * acc;
    }
}

// ---------- xp: im2col of match, pixel-major [b*NPIX][KDIM] bf16 (K contiguous) ----------
__global__ void k_xp(const float* __restrict__ match, unsigned short* __restrict__ xp) {
    int p = blockIdx.x;                  // b*NPIX + pix
    int b = p / NPIX, pix = p % NPIX;
    int h = pix / Wn, w = pix % Wn;
    for (int k = threadIdx.x; k < KDIM; k += blockDim.x) {
        int c = k / 9, r = k % 9, i = r / 3, j = r % 3;
        int hh = h + i - 1, ww = w + j - 1;
        float v = 0.f;
        if (hh >= 0 && hh < Hn && ww >= 0 && ww < Wn)
            v = match[((size_t)(b * CRn + c)) * (Hn * Wn) + hh * Wn + ww];
        xp[(size_t)p * KDIM + k] = f2bf(v);
    }
}

// ---------- wk: patches of ref + per-patch L2 normalize, [b*Ln][KDIM] bf16 ----------
__global__ void k_wk(const float* __restrict__ ref, unsigned short* __restrict__ wk) {
    __shared__ float vals[KDIM];
    __shared__ float red[256];
    int p = blockIdx.x;                  // b*Ln + l
    int b = p / Ln, l = p % Ln;
    int lh = l / LHn, lw = l % LHn;
    float ss = 0.f;
    for (int k = threadIdx.x; k < KDIM; k += blockDim.x) {
        int c = k / 9, r = k % 9, i = r / 3, j = r % 3;
        int hh = lh + i - 1, ww = lw + j - 1;
        float v = 0.f;
        if (hh >= 0 && hh < LHn && ww >= 0 && ww < LHn)
            v = ref[((size_t)(b * CRn + c)) * (LHn * LHn) + hh * LHn + ww];
        vals[k] = v; ss += v * v;
    }
    red[threadIdx.x] = ss;
    __syncthreads();
    for (int s = 128; s > 0; s >>= 1) {
        if (threadIdx.x < s) red[threadIdx.x] += red[threadIdx.x + s];
        __syncthreads();
    }
    float scale = 1.f / fmaxf(sqrtf(red[0]), 1e-4f);
    for (int k = threadIdx.x; k < KDIM; k += blockDim.x)
        wk[(size_t)p * KDIM + k] = f2bf(vals[k] * scale);
}

// ---------- A_tap: per-tap raw_w slice [b][81 taps][c][l] bf16 (l contiguous) ----------
__global__ void k_atap(const float* __restrict__ embed, unsigned short* __restrict__ atap) {
    int idx = blockIdx.x;                // ((b*81 + tap)*Cn + c)
    int c = idx % Cn; int bt = idx / Cn;
    int tap = bt % 81; int b = bt / 81;
    int ki = tap / 9, kj = tap % 9;
    const float* e = embed + (size_t)(b * Cn + c) * (Hn * Wn);
    unsigned short* dst = atap + (size_t)idx * Ln;
    for (int l = threadIdx.x; l < Ln; l += blockDim.x) {
        int lh = l / LHn, lw = l % LHn;
        int y = 3 * lh - 3 + ki, x = 3 * lw - 3 + kj;   // kernel=9, stride=3, SAME pad 3
        float v = 0.f;
        if (y >= 0 && y < Hn && x >= 0 && x < Wn) v = e[y * Wn + x];
        dst[l] = f2bf(v);
    }
}

// ---------- score GEMM: D[l,pix] = wk[l,:] . xp[pix,:]  (bf16 WMMA, f32 acc) ----------
// 8 waves = 4(M) x 2(N); wave tile 32x32 (2x2 frag block); macro tile 128 L x 64 pix.
// 4 WMMA per k-step per wave from 2 A-frags + 2 B-frags -> 16 flop/byte from L2.
__global__ void k_score_gemm(const unsigned short* __restrict__ wk,
                             const unsigned short* __restrict__ xp,
                             float* __restrict__ scoreT) {
    int gid = blockIdx.x;
    int pblk = gid % 144; int t = gid / 144;   // NPIX/64 = 144
    int lblk = t % 8;     int b = t / 8;       // Ln/128 = 8
    int wv = threadIdx.x >> 5, lane = threadIdx.x & 31;
    int mw = wv & 3, nw = wv >> 2;
    int lbase = lblk * 128 + mw * 32;
    int pbase = pblk * 64 + nw * 32;
    const unsigned short* A  = wk + ((size_t)b * Ln   + lbase) * KDIM;
    const unsigned short* Bt = xp + ((size_t)b * NPIX + pbase) * KDIM;
    int m = lane & 15, khalf = lane >> 4;
    const uint4* a0 = (const uint4*)(A  + (size_t)m        * KDIM + khalf * 16);
    const uint4* a1 = (const uint4*)(A  + (size_t)(m + 16) * KDIM + khalf * 16);
    const uint4* b0 = (const uint4*)(Bt + (size_t)m        * KDIM + khalf * 16);
    const uint4* b1 = (const uint4*)(Bt + (size_t)(m + 16) * KDIM + khalf * 16);
    v8f_t acc00 = {}, acc01 = {}, acc10 = {}, acc11 = {};
    for (int ks = 0; ks < KDIM / 32; ++ks) {      // 18 k-steps of 32
        FragU fa0, fa1, fb0, fb1;
        fa0.u[0] = a0[ks * 4]; fa0.u[1] = a0[ks * 4 + 1];
        fa1.u[0] = a1[ks * 4]; fa1.u[1] = a1[ks * 4 + 1];
        fb0.u[0] = b0[ks * 4]; fb0.u[1] = b0[ks * 4 + 1];
        fb1.u[0] = b1[ks * 4]; fb1.u[1] = b1[ks * 4 + 1];
        acc00 = __builtin_amdgcn_wmma_f32_16x16x32_bf16(false, fa0.v, false, fb0.v, (short)0, acc00, false, false);
        acc01 = __builtin_amdgcn_wmma_f32_16x16x32_bf16(false, fa0.v, false, fb1.v, (short)0, acc01, false, false);
        acc10 = __builtin_amdgcn_wmma_f32_16x16x32_bf16(false, fa1.v, false, fb0.v, (short)0, acc10, false, false);
        acc11 = __builtin_amdgcn_wmma_f32_16x16x32_bf16(false, fa1.v, false, fb1.v, (short)0, acc11, false, false);
    }
    // store transposed: scoreT[b][pix][l]
    int n = lane & 15, mo = (lane >> 4) * 8;
    v8f_t accs[2][2] = {{acc00, acc01}, {acc10, acc11}};
    #pragma unroll
    for (int i = 0; i < 2; ++i) {
        #pragma unroll
        for (int j = 0; j < 2; ++j) {
            float* out = scoreT + ((size_t)b * NPIX + (pbase + j * 16 + n)) * Ln
                       + lbase + i * 16 + mo;
            #pragma unroll
            for (int r = 0; r < 8; ++r) out[r] = accs[i][j][r];
        }
    }
}

// ---------- softmax over L per (b,pix): one wave per row of 1024 ----------
__global__ void k_softmax(const float* __restrict__ scoreT, unsigned short* __restrict__ sbf) {
    int wv = threadIdx.x >> 5, lane = threadIdx.x & 31;
    size_t row = (size_t)blockIdx.x * 8 + wv;          // over B*NPIX
    const float* src = scoreT + row * Ln;
    unsigned short* dst = sbf + row * Ln;
    float v[32];
    float mx = -1e30f;
    #pragma unroll
    for (int i = 0; i < 32; ++i) { v[i] = 10.f * src[lane + i * 32]; mx = fmaxf(mx, v[i]); }
    for (int d = 16; d; d >>= 1) mx = fmaxf(mx, __shfl_xor(mx, d, 32));
    float sum = 0.f;
    #pragma unroll
    for (int i = 0; i < 32; ++i) { v[i] = __expf(v[i] - mx); sum += v[i]; }
    for (int d = 16; d; d >>= 1) sum += __shfl_xor(sum, d, 32);
    float inv = 1.f / sum;
    #pragma unroll
    for (int i = 0; i < 32; ++i) dst[lane + i * 32] = f2bf(v[i] * inv);
}

// ---------- deconv: 9 tap-GEMMs accumulated; wave tile = 32 ch x 16 out pixels ----------
// 8 waves = 4(M: 128 ch) x 2(N: 32 px). Same oh & ow-phase across block -> tap uniform.
// 2 WMMA per k-step per wave from 2 A-frags + 1 B-frag.
__global__ void k_deconv(const unsigned short* __restrict__ atap,
                         const unsigned short* __restrict__ sbf,
                         const unsigned short* __restrict__ zrow,
                         float* __restrict__ out) {
    int gid = blockIdx.x;
    int tile = gid % 3; gid /= 3;               // 3 x 32 px = 96 positions per (oh,rw)
    int rw   = gid % 3; gid /= 3;
    int oh   = gid % OHW; int b = gid / OHW;
    int wv = threadIdx.x >> 5, lane = threadIdx.x & 31;
    int mw = wv & 3, nw = wv >> 2;
    int cbase = mw * 32;
    int m = lane & 15, khalf = lane >> 4;
    int tpos = 32 * tile + nw * 16 + m;         // this lane's pixel slot (as B row n)
    int rh = oh % 3;
    v8f_t acc0 = {}, acc1 = {};
    for (int ai = 0; ai < 3; ++ai) {
        int ki = rh + 3 * ai;
        int h = (oh + 3 - ki) / 3;              // conv_transpose pad = SCALE = 3
        if (h < 0 || h >= Hn) continue;
        for (int aj = 0; aj < 3; ++aj) {
            int kj = rw + 3 * aj;
            int wcol = (rw + 3 - kj) / 3 + tpos;   // per-lane source column
            const unsigned short* A = atap +
                ((size_t)((b * 81 + ki * 9 + kj) * Cn) + cbase) * Ln;
            const uint4* a0 = (const uint4*)(A + (size_t)m        * Ln + khalf * 16);
            const uint4* a1 = (const uint4*)(A + (size_t)(m + 16) * Ln + khalf * 16);
            const unsigned short* Brow = (wcol >= 0 && wcol < Wn)
                ? (sbf + ((size_t)b * NPIX + h * Wn + wcol) * Ln)
                : zrow;
            const uint4* b0 = (const uint4*)(Brow + khalf * 16);
            for (int ks = 0; ks < Ln / 32; ++ks) {  // 32 k-steps per tap
                FragU fa0, fa1, fb;
                fa0.u[0] = a0[ks * 4]; fa0.u[1] = a0[ks * 4 + 1];
                fa1.u[0] = a1[ks * 4]; fa1.u[1] = a1[ks * 4 + 1];
                fb.u[0]  = b0[ks * 4]; fb.u[1]  = b0[ks * 4 + 1];
                acc0 = __builtin_amdgcn_wmma_f32_16x16x32_bf16(false, fa0.v, false, fb.v, (short)0, acc0, false, false);
                acc1 = __builtin_amdgcn_wmma_f32_16x16x32_bf16(false, fa1.v, false, fb.v, (short)0, acc1, false, false);
            }
        }
    }
    int n = lane & 15, mo = (lane >> 4) * 8;
    int ow = rw + 3 * (32 * tile + nw * 16 + n);
    #pragma unroll
    for (int r = 0; r < 8; ++r) {
        out[(((size_t)b * Cn + cbase +      mo + r) * OHW + oh) * OHW + ow] = acc0[r] * (1.f / 6.f);
        out[(((size_t)b * Cn + cbase + 16 + mo + r) * OHW + oh) * OHW + ow] = acc1[r] * (1.f / 6.f);
    }
}

extern "C" void kernel_launch(void* const* d_in, const int* in_sizes, int n_in,
                              void* d_out, int out_size, void* d_ws, size_t ws_size,
                              hipStream_t stream) {
    const float* x     = (const float*)d_in[0];
    const float* w_asm = (const float*)d_in[1];
    const float* b_asm = (const float*)d_in[2];
    const float* a_asm = (const float*)d_in[3];
    const float* w_m1  = (const float*)d_in[4];
    const float* b_m1  = (const float*)d_in[5];
    const float* a_m1  = (const float*)d_in[6];
    const float* w_m2  = (const float*)d_in[7];
    const float* b_m2  = (const float*)d_in[8];
    const float* a_m2  = (const float*)d_in[9];
    float* out = (float*)d_out;

    char* ws = (char*)d_ws;
    constexpr size_t off_embed  = 0;                       // f32 [4][128][96][96]  18.9 MB
    constexpr size_t off_match  = off_embed  + 18874368;   // f32 [4][64][96][96]    9.4 MB
    constexpr size_t off_ref    = off_match  + 9437184;    // f32 [4][64][32][32]    1.0 MB
    constexpr size_t off_wk     = off_ref    + 1048576;    // bf16 [4*1024][576]     4.7 MB
    constexpr size_t off_xp     = off_wk     + 4718592;    // bf16 [4*9216][576]    42.5 MB
    constexpr size_t off_scoref = off_xp     + 42467328;   // f32 [4*9216][1024]   151.0 MB
    constexpr size_t off_scoreb = off_scoref + 150994944;  // bf16 [4*9216][1024]   75.5 MB
    constexpr size_t off_atap   = off_scoreb + 75497472;   // bf16 [4][81][128][1024] 84.9 MB
    constexpr size_t off_zero   = off_atap   + 84934656;   // bf16 [1024] zeros

    float* embed = (float*)(ws + off_embed);
    float* match = (float*)(ws + off_match);
    float* ref   = (float*)(ws + off_ref);
    unsigned short* wk     = (unsigned short*)(ws + off_wk);
    unsigned short* xp     = (unsigned short*)(ws + off_xp);
    float* scoreT          = (float*)(ws + off_scoref);
    unsigned short* sbf    = (unsigned short*)(ws + off_scoreb);
    unsigned short* atap   = (unsigned short*)(ws + off_atap);
    unsigned short* zrow   = (unsigned short*)(ws + off_zero);

    k_zero<<<4, 256, 0, stream>>>(zrow);
    k_embed_match<<<Bn * Hn * Wn, 128, 0, stream>>>(x, w_asm, b_asm, a_asm,
                                                    w_m1, b_m1, a_m1, embed, match);
    k_ref<<<Bn * LHn * LHn, 128, 0, stream>>>(x, w_m2, b_m2, a_m2, ref);
    k_xp<<<Bn * NPIX, 192, 0, stream>>>(match, xp);
    k_wk<<<Bn * Ln, 256, 0, stream>>>(ref, wk);
    k_atap<<<Bn * 81 * Cn, 256, 0, stream>>>(embed, atap);
    k_score_gemm<<<Bn * 8 * 144, 256, 0, stream>>>(wk, xp, scoreT);
    k_softmax<<<Bn * NPIX / 8, 256, 0, stream>>>(scoreT, sbf);
    k_deconv<<<Bn * OHW * 3 * 3, 256, 0, stream>>>(atap, sbf, zrow, out);
}
